// SoftPixelCNN_36094905155950
// MI455X (gfx1250) — compile-verified, matrix operand order checked by reading, not run on hardware
//
#include <hip/hip_runtime.h>

typedef __attribute__((ext_vector_type(16))) _Float16 v16h;
typedef __attribute__((ext_vector_type(8)))  _Float16 v8h;
typedef __attribute__((ext_vector_type(8)))  float    v8f;
typedef __attribute__((ext_vector_type(4)))  float    v4f;

namespace {
constexpr int   kD  = 4;    // coordinate dims
constexpr int   kF  = 64;   // features
constexpr int   kK  = 32;   // neighbours == WMMA K for 16x16x32_f16
constexpr int   kO  = 9;    // offsets (origin + 8 axis points), padded to M=16
constexpr int   kOF = kO * kF;          // 576 output floats per vertex
constexpr float kGaussExp = 10.0f;
constexpr float kLnInvK   = -3.4657359027997265f;   // ln(1/32): folds mean-over-K into exp

constexpr int kWavesPerBlock = 8;
constexpr int kThreads = kWavesPerBlock * 32;
constexpr int kAHalfs = 16 * kK;        // 16x32 f16 A tile  (1 KB)
constexpr int kBHalfs = kF * kK;        // 64x32 f16 Bt tile (4 KB), [f][k]
constexpr int kWaveHalfs = kAHalfs + kBHalfs;   // 5 KB/wave; C (16x64 f32 = 4 KB) reuses it
}

__global__ __launch_bounds__(kThreads)
void spcnn_wmma_kernel(const float* __restrict__ coords,
                       const float* __restrict__ feats,
                       const int*   __restrict__ nidx,
                       const float* __restrict__ lscale,
                       float*       __restrict__ out,
                       int V)
{
    __shared__ __attribute__((aligned(16)))
        _Float16 lds[kWavesPerBlock * kWaveHalfs];

    const int wave = threadIdx.x >> 5;
    const int lane = threadIdx.x & 31;
    _Float16* __restrict__ Aw = lds + wave * kWaveHalfs;
    _Float16* __restrict__ Bw = Aw + kAHalfs;

    const int v = blockIdx.x * kWavesPerBlock + wave;   // uniform within wave
    if (v >= V) return;                                  // wave-uniform branch

    // zero-pad A rows o = 9..15 (each lane owns one k-column)
    #pragma unroll
    for (int o = kO; o < 16; ++o) Aw[o * kK + lane] = (_Float16)0.0f;

    const float s = -kGaussExp * lscale[0];

    // lane k = this lane's neighbour
    const int    ni = nidx[(long)v * kK + lane];
    const float4 cv = *(const float4*)(coords + (long)v  * kD);
    const float4 cn = *(const float4*)(coords + (long)ni * kD);
    const float d0 = cv.x - cn.x;
    const float d1 = cv.y - cn.y;
    const float d2 = cv.z - cn.z;
    const float d3 = cv.w - cn.w;
    const float base = d0*d0 + d1*d1 + d2*d2 + d3*d3;

    // offsets (meshgrid 'xy' filtered order):
    // 0:(0,0,0,0) 1:(0,-1,0,0) 2:(-1,0,0,0) 3:(0,0,-1,0) 4:(0,0,0,-1)
    // 5:(0,0,0,1) 6:(0,0,1,0)  7:(1,0,0,0)  8:(0,1,0,0)
    float dq[kO];
    dq[0] = base;
    dq[1] = base - 2.0f*d1 + 1.0f;
    dq[2] = base - 2.0f*d0 + 1.0f;
    dq[3] = base - 2.0f*d2 + 1.0f;
    dq[4] = base - 2.0f*d3 + 1.0f;
    dq[5] = base + 2.0f*d3 + 1.0f;
    dq[6] = base + 2.0f*d2 + 1.0f;
    dq[7] = base + 2.0f*d0 + 1.0f;
    dq[8] = base + 2.0f*d1 + 1.0f;

    // weight = exp(s*d)/K = exp(s*d + ln(1/K)) -> one fma feeding v_exp_f32
    #pragma unroll
    for (int o = 0; o < kO; ++o)
        Aw[o * kK + lane] = (_Float16)__expf(__builtin_fmaf(s, dq[o], kLnInvK));

    // gather this neighbour's 64 features, f16-convert, store transposed Bt[f][k]
    const float* __restrict__ frow = feats + (long)ni * kF;
    #pragma unroll
    for (int c = 0; c < kF; c += 4) {
        const float4 q = *(const float4*)(frow + c);
        Bw[(c + 0) * kK + lane] = (_Float16)q.x;
        Bw[(c + 1) * kK + lane] = (_Float16)q.y;
        Bw[(c + 2) * kK + lane] = (_Float16)q.z;
        Bw[(c + 3) * kK + lane] = (_Float16)q.w;
    }

    // intra-wave LDS transpose barrier: DS ops of a wave complete in order
    asm volatile("s_wait_dscnt 0x0" ::: "memory");

    // A fragment (16-bit A 16x32 layout): lane half h, row r:
    //   a[0..7]  = A[r][h*8 + 0..7],  a[8..15] = A[r][16 + h*8 + 0..7]
    const int half = lane >> 4;
    const int row  = lane & 15;
    v16h a;
    {
        const v8h a0 = *(const v8h*)(Aw + row * kK + half * 8);
        const v8h a1 = *(const v8h*)(Aw + row * kK + 16 + half * 8);
        #pragma unroll
        for (int j = 0; j < 8; ++j) { a[j] = a0[j]; a[8 + j] = a1[j]; }
    }

    // B fragments for all 4 N-tiles, then all 4 WMMAs (EXEC all-ones throughout)
    v8f acc[4];
    #pragma unroll
    for (int t = 0; t < 4; ++t) {
        const _Float16* bp = Bw + (t * 16 + row) * kK + half * 8;
        const v8h b0 = *(const v8h*)bp;
        const v8h b1 = *(const v8h*)(bp + 16);
        v16h b;
        #pragma unroll
        for (int j = 0; j < 8; ++j) { b[j] = b0[j]; b[8 + j] = b1[j]; }

        v8f z = {};
        acc[t] = __builtin_amdgcn_wmma_f32_16x16x32_f16(
                     /*neg_a=*/false, a, /*neg_b=*/false, b,
                     /*c_mod=*/(short)0, z, /*reuse_a=*/false, /*reuse_b=*/false);
    }

    // Branch-free epilogue: dump C tiles (16x64 f32) into the wave's LDS slice
    // (B fragments already consumed; DS ops are in-order within a wave), then
    // stream the 576 valid floats out as coalesced non-temporal b128 stores.
    float* __restrict__ Cld = (float*)Aw;       // 4 KB reuse of A+B slice
    #pragma unroll
    for (int t = 0; t < 4; ++t)
        #pragma unroll
        for (int j = 0; j < 8; ++j)             // M = half*8 + j, N = t*16 + row
            Cld[(half * 8 + j) * kF + t * 16 + row] = acc[t][j];

    asm volatile("s_wait_dscnt 0x0" ::: "memory");

    float* __restrict__ orow = out + (long)v * kOF;     // 16B-aligned (576*4 = 2304)
    #pragma unroll
    for (int c = 0; c < 4; ++c) {               // 4 x 512B full-wave chunks
        const v4f q = *(const v4f*)(Cld + c * 128 + lane * 4);
        __builtin_nontemporal_store(q, (v4f*)(orow + c * 128 + lane * 4));
    }
    if (lane < 16) {                            // 64-float tail (row 8 of C)
        const v4f q = *(const v4f*)(Cld + 512 + lane * 4);
        __builtin_nontemporal_store(q, (v4f*)(orow + 512 + lane * 4));
    }
}

extern "C" void kernel_launch(void* const* d_in, const int* in_sizes, int n_in,
                              void* d_out, int out_size, void* d_ws, size_t ws_size,
                              hipStream_t stream) {
    (void)n_in; (void)out_size; (void)d_ws; (void)ws_size;
    const float* coords = (const float*)d_in[0];   // (V,4)  f32
    const float* feats  = (const float*)d_in[1];   // (V,64) f32
    // d_in[2] = distsq: unused on the inference path (stop_gradient -> EMA only)
    const int*   nidx   = (const int*)d_in[3];     // (V,32) i32
    const float* ls     = (const float*)d_in[4];   // (1,)   f32
    float* out = (float*)d_out;                    // (V,576) f32

    const int V = in_sizes[0] / kD;
    const int blocks = (V + kWavesPerBlock - 1) / kWavesPerBlock;
    spcnn_wmma_kernel<<<blocks, kThreads, 0, stream>>>(coords, feats, nidx, ls, out, V);
}
